// PriorFocalModifierLoss_33612414058942
// MI455X (gfx1250) — matrix operand order repaired and checked by reading, not compiled, not run on hardware
//
#include <hip/hip_runtime.h>
#include <hip/hip_bf16.h>

// ---------------------------------------------------------------------------
// PriorFocalModifierLoss on MI455X (gfx1250, wave32, WMMA)
//
// Dominant work: att_raw = y @ M  (16384x1000 @ 1000x1000, ~32.8 GFLOP).
// HBM floor: x (65.5MB) + y (65.5MB, read ONCE via LDS reuse) + co (~8MB)
// ~= 135MB -> ~6us @ 23.3 TB/s. The GEMM then only needs a few TFLOP/s, so
// v_wmma_f32_16x16x32_f16 (y exact in f16, M small positive fractions, f32
// accumulate) keeps it memory-bound, with the focal-loss epilogue fused so
// the BxC att_raw matrix never leaves registers.
//
// Normalizer trick: row_sum(att_raw)[b] = sum_c y[b,c] * rowsumM[c], so the
// padded B matrix carries rowsumM as virtual column n=1000; the GEMM makes
// the per-row normalizer for free.
//
// LDS reuse: the 16x1024 Y tile is staged to LDS (f16, 33KB) exactly once and
// serves BOTH as the WMMA A-operand source and as the epilogue's y values,
// halving Y's HBM traffic and eliminating all K-loop barriers.
// ---------------------------------------------------------------------------

typedef __attribute__((ext_vector_type(16))) _Float16 v16h;
typedef __attribute__((ext_vector_type(8)))  float    v8f;

#define C_DIM 1000
#define KP    1024          // padded K (classes, A/B depth)
#define NP    1024          // padded N (classes + rowsum column + pad)
#define YPITCH 1032         // Ys row pitch in halfs (2064B: 16B-aligned, bank-skewed)

__global__ void zero_out_kernel(float* out) {
    if (threadIdx.x == 0) out[0] = 0.0f;
}

// colsum[d] = sum_c co[c][d]   (column sums for M normalization)
__global__ void colsum_kernel(const float* __restrict__ co,
                              float* __restrict__ colsum) {
    int d = blockIdx.x * blockDim.x + threadIdx.x;
    if (d < C_DIM) {
        float s = 0.0f;
        for (int c = 0; c < C_DIM; ++c) s += co[(size_t)c * C_DIM + d];
        colsum[d] = s;
    }
}

// BT[n][k] = M[k][n] = co[k][n] / colsum[n], stored f16, K-contiguous rows
// so each WMMA B fragment is one 32B contiguous load. Zero padding elsewhere.
__global__ void build_bt_kernel(const float* __restrict__ co,
                                const float* __restrict__ colsum,
                                _Float16* __restrict__ bt) {
    const int d = blockIdx.x;                       // n index, 0..1023
    const float inv = (d < C_DIM) ? (1.0f / colsum[d]) : 0.0f;
    for (int c = threadIdx.x; c < KP; c += blockDim.x) {
        float v = 0.0f;
        if (d < C_DIM && c < C_DIM) v = co[(size_t)c * C_DIM + d] * inv;
        bt[(size_t)d * KP + c] = (_Float16)v;
    }
}

// Virtual column n=1000: s[c] = sum_d M[c][d], using the f16-rounded M values
// so the normalizer matches what the WMMA GEMM actually accumulates.
__global__ void rowsum_kernel(const float* __restrict__ co,
                              const float* __restrict__ colsum,
                              _Float16* __restrict__ bt) {
    __shared__ float red[256];
    const int c = blockIdx.x;                       // 0..999
    float s = 0.0f;
    for (int d = threadIdx.x; d < C_DIM; d += blockDim.x) {
        float m = co[(size_t)c * C_DIM + d] / colsum[d];
        s += (float)(_Float16)m;                    // round like BT entries
    }
    red[threadIdx.x] = s;
    __syncthreads();
    for (int st = 128; st > 0; st >>= 1) {
        if (threadIdx.x < st) red[threadIdx.x] += red[threadIdx.x + st];
        __syncthreads();
    }
    if (threadIdx.x == 0) bt[(size_t)C_DIM * KP + c] = (_Float16)red[0];
}

// Fused GEMM + focal-loss epilogue.
// Block: 256 threads = 8 waves; 16 rows x 1024 cols per block.
// Wave w owns columns [w*128, w*128+128) -> 8 accumulator tiles (v8f each).
__global__ void __launch_bounds__(256)
gemm_loss_kernel(const float* __restrict__ X,
                 const float* __restrict__ Y,
                 const _Float16* __restrict__ BT,
                 const float* __restrict__ W,
                 float* __restrict__ out) {
    __shared__ _Float16 Ys[16 * YPITCH];   // full 16x1024 y tile, f16 (33KB)
    __shared__ float rowsum_s[16];
    __shared__ float red[256];

    const int tid  = threadIdx.x;
    const int lane = tid & 31;
    const int wave = tid >> 5;
    const int r    = lane & 15;        // column-within-tile (B/C/D layout)
    const int g    = lane >> 4;        // lane-group (K/M half select)
    const int row0 = blockIdx.x * 16;

    // ---- Stage the whole 16x1024 Y tile once (coalesced, f32 -> f16) ----
    for (int idx = tid; idx < 16 * 1024; idx += 256) {
        const int row = idx >> 10;
        const int col = idx & 1023;
        float v = (col < C_DIM) ? Y[(size_t)(row0 + row) * C_DIM + col] : 0.0f;
        Ys[row * YPITCH + col] = (_Float16)v;
    }
    __syncthreads();                   // only barrier before the K loop

    v8f c[8] = {};

    for (int kk = 0; kk < KP; kk += 32) {
        // A fragment, contiguous K order (B at line-matched order):
        //   halves 0..7  -> K = kk + 16g + h
        //   halves 8..15 -> K = kk + 16g + 8 + h
        v16h a;
        {
            const _Float16* ap = &Ys[r * YPITCH + kk + 16 * g];
            #pragma unroll
            for (int h = 0; h < 8; ++h) {
                a[h]     = ap[h];
                a[h + 8] = ap[8 + h];
            }
        }

        // Load all 8 B fragments first so the loads clause together and the
        // WMMAs below overlap with in-flight loads (no full-drain waits).
        v16h bf[8];
        #pragma unroll
        for (int t = 0; t < 8; ++t) {
            const int n = wave * 128 + t * 16 + r;
            bf[t] = *(const v16h*)&BT[(size_t)n * KP + kk + 16 * g];
        }
        #pragma unroll
        for (int t = 0; t < 8; ++t) {
            c[t] = __builtin_amdgcn_wmma_f32_16x16x32_f16(
                false, a, false, bf[t], (short)0, c[t], false, false);
        }
    }

    // Column n=1000 (the row-sum normalizer) lives in wave 7, tile 6, r==8.
    // C/D layout: lane r -> N, VGPR v -> M = v + 8g.
    if (wave == 7 && r == 8) {
        #pragma unroll
        for (int v = 0; v < 8; ++v) rowsum_s[v + 8 * g] = c[6][v];
    }
    __syncthreads();

    // Fused focal-loss epilogue straight from register accumulators.
    // y comes from the LDS tile (already staged) -> Y is read once from HBM.
    float acc = 0.0f;
    #pragma unroll
    for (int t = 0; t < 8; ++t) {
        const int n = wave * 128 + t * 16 + r;
        if (n < C_DIM) {
            const float w       = W[n];
            const float osg_neg = 3.0f + w;          // GAMMA_NEG + weight
            #pragma unroll
            for (int v = 0; v < 8; ++v) {
                const int m    = v + 8 * g;
                const float rs  = rowsum_s[m];
                const float att = c[t][v] / rs;
                const float x = X[(size_t)(row0 + m) * C_DIM + n];
                const float y = (float)Ys[m * YPITCH + n];

                const float xsig   = 1.0f / (1.0f + expf(-x));
                const float xs_pos = xsig * (1.0f - att);          // GAMMA_CLASS_POS - att
                float xs_neg = fminf(1.0f - xsig + 0.05f, 1.0f);   // + CLIP
                xs_neg = fminf((att == 0.0f) ? xs_neg : xs_neg * 1.2f, 1.0f);

                const float lp = y * logf(fmaxf(xs_pos, 1e-8f));
                const float ln = (1.0f - y) * logf(fmaxf(xs_neg, 1e-8f));
                const float losse = (lp + ln) * w;

                const float pt  = xs_pos * y + xs_neg * (1.0f - y);
                const float osg = y + osg_neg * (1.0f - y);        // GAMMA_POS*y + ...
                const float osw = powf(fmaxf(1.0f - pt, 0.0f), osg);
                acc += losse * osw;
            }
        }
    }

    red[tid] = acc;
    __syncthreads();
    for (int st = 128; st > 0; st >>= 1) {
        if (tid < st) red[tid] += red[tid + st];
        __syncthreads();
    }
    if (tid == 0) atomicAdd(out, -red[0]);   // reference returns -loss.sum()
}

extern "C" void kernel_launch(void* const* d_in, const int* in_sizes, int n_in,
                              void* d_out, int out_size, void* d_ws, size_t ws_size,
                              hipStream_t stream) {
    const float* x  = (const float*)d_in[0];
    const float* y  = (const float*)d_in[1];
    const float* co = (const float*)d_in[2];
    const float* w  = (const float*)d_in[3];
    float* out = (float*)d_out;

    // Workspace: [0,4KB) colsum, [4KB, 4KB+2MB) f16 B^T (1024x1024).
    float*     colsum = (float*)d_ws;
    _Float16*  bt     = (_Float16*)((char*)d_ws + 4096);

    const int nrows = in_sizes[0] / C_DIM;   // B = 16384

    zero_out_kernel<<<1, 64, 0, stream>>>(out);
    colsum_kernel<<<(C_DIM + 255) / 256, 256, 0, stream>>>(co, colsum);
    build_bt_kernel<<<NP, 256, 0, stream>>>(co, colsum, bt);
    rowsum_kernel<<<C_DIM, 256, 0, stream>>>(co, colsum, bt);
    gemm_loss_kernel<<<nrows / 16, 256, 0, stream>>>(x, y, bt, w, out);
}
